// MAGClassifier_7885559955811
// MI455X (gfx1250) — compile-verified, matrix operand order checked by reading, not run on hardware
//
#include <hip/hip_runtime.h>

// ---------------------------------------------------------------------------
// Problem constants (from reference)
// ---------------------------------------------------------------------------
#define GG   32
#define NN   128
#define EE   512
#define DD   256
#define HEADS 8
#define DH   32          // DD / HEADS
#define HIDC 128
#define NROW (GG * EE)   // 16384 edge tokens

typedef __attribute__((ext_vector_type(16))) __bf16 v16bf;
typedef __attribute__((ext_vector_type(8)))  float  v8f;

#if defined(__has_builtin)
#if __has_builtin(__builtin_amdgcn_sched_barrier)
#define SCHED_BARRIER() __builtin_amdgcn_sched_barrier(0)
#endif
#endif
#ifndef SCHED_BARRIER
#define SCHED_BARRIER() ((void)0)
#endif

// ---------------------------------------------------------------------------
// Device helpers
// ---------------------------------------------------------------------------
__device__ __forceinline__ v8f wmma_bf16(v16bf a, v16bf b, v8f c) {
    return __builtin_amdgcn_wmma_f32_16x16x32_bf16(false, a, false, b, (short)0, c,
                                                   false, false);
}

// Load a 16x32 bf16 fragment (A layout; also valid for B-operand columns when
// the 32 K-elements of each column are contiguous in memory).
// lane 0..15  : row (lane),    K halves {0..7, 16..23}
// lane 16..31 : row (lane-16), K halves {8..15, 24..31}
__device__ __forceinline__ v16bf load_frag(const __bf16* base, int ld, int lane) {
    const int r  = lane & 15;
    const int ko = (lane & 16) ? 8 : 0;
    const __bf16* p = base + (size_t)r * ld + ko;
    v16bf f;
#pragma unroll
    for (int i = 0; i < 8; ++i) f[i] = p[i];
#pragma unroll
    for (int i = 0; i < 8; ++i) f[i + 8] = p[i + 16];
    return f;
}

__device__ __forceinline__ v16bf load_frag_guard(const __bf16* base, int ld, int lane,
                                                 int rows_valid) {
    const int r = lane & 15;
    v16bf f;
    if (r < rows_valid) {
        f = load_frag(base, ld, lane);
    } else {
#pragma unroll
        for (int i = 0; i < 16; ++i) f[i] = (__bf16)0.0f;
    }
    return f;
}

// ---------------------------------------------------------------------------
// Weight transpose + fp32 -> bf16  ( W (K x N) -> Wt (N x K) )
// ---------------------------------------------------------------------------
__global__ void transpose_w_kernel(const float* __restrict__ W,
                                   unsigned short* __restrict__ Wt_u,
                                   int K, int N) {
    __bf16* Wt = (__bf16*)Wt_u;
    int idx = blockIdx.x * blockDim.x + threadIdx.x;
    if (idx >= K * N) return;
    int n = idx / K, k = idx % K;
    Wt[idx] = (__bf16)W[(size_t)k * N + n];
}

// ---------------------------------------------------------------------------
// Small fp32 projections:  out (M x 128) = X (M x K) @ W (K x 128) + b
// ---------------------------------------------------------------------------
__global__ void proj_kernel(const float* __restrict__ X, const float* __restrict__ W,
                            const float* __restrict__ b, float* __restrict__ out,
                            int M, int K) {
    int idx = blockIdx.x * blockDim.x + threadIdx.x;
    if (idx >= M * 128) return;
    int r = idx >> 7, c = idx & 127;
    float acc = b[c];
    for (int k = 0; k < K; ++k) acc += X[(size_t)r * K + k] * W[(size_t)k * 128 + c];
    out[idx] = acc;
}

// ---------------------------------------------------------------------------
// edge_repr assembly -> er bf16 (NROW x 256)
// ---------------------------------------------------------------------------
__global__ void assemble_kernel(const float* __restrict__ nodef,
                                const float* __restrict__ edgef,
                                const int* __restrict__ ei,
                                unsigned short* __restrict__ erB_u) {
    __bf16* erB = (__bf16*)erB_u;
    int idx = blockIdx.x * blockDim.x + threadIdx.x;
    if (idx >= NROW * DD) return;
    int r = idx >> 8, c = idx & 255;
    int node = (c < 128) ? ei[r] : ei[NROW + r];  // src / dst (global node ids)
    int cc = c & 127;
    float v = nodef[(size_t)node * 128 + cc] + edgef[(size_t)r * 128 + cc];
    erB[idx] = (__bf16)v;
}

// ---------------------------------------------------------------------------
// adjacency mask (G x E x E) u8
// ---------------------------------------------------------------------------
__global__ void mask_kernel(const int* __restrict__ ei, unsigned char* __restrict__ mask) {
    __shared__ int s[EE];
    __shared__ int d[EE];
    int g = blockIdx.x;
    for (int i = threadIdx.x; i < EE; i += blockDim.x) {
        s[i] = ei[g * EE + i];
        d[i] = ei[NROW + g * EE + i];
    }
    __syncthreads();
    unsigned char* mg = mask + (size_t)g * EE * EE;
    for (int idx = threadIdx.x; idx < EE * EE; idx += blockDim.x) {
        int i = idx >> 9, j = idx & 511;
        int si = s[i], di = d[i], sj = s[j], dj = d[j];
        mg[idx] = (si == sj || si == dj || di == sj || di == dj) ? 1 : 0;
    }
}

// ---------------------------------------------------------------------------
// WMMA bf16 GEMM:  out (M x N) = A (M x K, bf16) @ Wt^T (Wt is N x K, bf16) + bias
// Each wave computes a 16x64 strip (4 accumulators, shared A-fragment).
// N must be a multiple of 64.  4 waves / block.
// sched_barrier keeps all 5 fragment loads of a k-step issued before the
// wmma group so waitcnt insertion can use partial waits (load/XDL overlap).
// ---------------------------------------------------------------------------
__global__ void __launch_bounds__(128, 1)
gemm_bf16_kernel(const unsigned short* __restrict__ A_u,
                 const unsigned short* __restrict__ Wt_u,
                 const float* __restrict__ bias,
                 float* __restrict__ outF,
                 unsigned short* __restrict__ outB_u,
                 int M, int N, int K, int relu) {
    const __bf16* A  = (const __bf16*)A_u;
    const __bf16* Wt = (const __bf16*)Wt_u;
    __bf16* outB = (__bf16*)outB_u;

    int lane = threadIdx.x & 31;
    int warp = threadIdx.x >> 5;
    int nsup   = N >> 6;               // 64-wide supertiles along N
    int mtiles = (M + 15) >> 4;
    int task = blockIdx.x * 4 + warp;
    if (task >= mtiles * nsup) return;
    int mt  = task / nsup;
    int nt4 = task % nsup;

    const __bf16* Abase = A + (size_t)(mt * 16) * K;
    const __bf16* B0 = Wt + (size_t)(nt4 * 64)      * K;
    const __bf16* B1 = Wt + (size_t)(nt4 * 64 + 16) * K;
    const __bf16* B2 = Wt + (size_t)(nt4 * 64 + 32) * K;
    const __bf16* B3 = Wt + (size_t)(nt4 * 64 + 48) * K;

    v8f c0 = {}, c1 = {}, c2 = {}, c3 = {};
    for (int k = 0; k < K; k += 32) {
        v16bf af  = load_frag(Abase + k, K, lane);
        v16bf bf0 = load_frag(B0 + k, K, lane);
        v16bf bf1 = load_frag(B1 + k, K, lane);
        v16bf bf2 = load_frag(B2 + k, K, lane);
        v16bf bf3 = load_frag(B3 + k, K, lane);
        SCHED_BARRIER();   // all loads above, all wmmas below
        c0 = wmma_bf16(af, bf0, c0);
        c1 = wmma_bf16(af, bf1, c1);
        c2 = wmma_bf16(af, bf2, c2);
        c3 = wmma_bf16(af, bf3, c3);
        SCHED_BARRIER();
    }

    int colb  = lane & 15;
    int rbase = mt * 16 + ((lane & 16) ? 8 : 0);
    v8f acc[4] = {c0, c1, c2, c3};
#pragma unroll
    for (int t = 0; t < 4; ++t) {
        int col = nt4 * 64 + t * 16 + colb;
        float bv = bias ? bias[col] : 0.0f;
#pragma unroll
        for (int j = 0; j < 8; ++j) {
            int row = rbase + j;
            if (row >= M) continue;
            float v = acc[t][j] + bv;
            if (relu) v = fmaxf(v, 0.0f);
            if (outF) outF[(size_t)row * N + col] = v;
            if (outB) outB[(size_t)row * N + col] = (__bf16)v;
        }
    }
}

// ---------------------------------------------------------------------------
// Flash attention (per graph g, head h).  Q: (G*Lq x 256), K/V: (G*Lk x 256),
// head h uses cols [h*32, h*32+32).  Online softmax, WMMA everywhere.
// Block = 4 waves; wave w handles query rows [qc*64 + w*16, +16).
// LDS: V transposed (32 x Lk bf16) + per-wave P staging (16x32 bf16).
// ---------------------------------------------------------------------------
__global__ void __launch_bounds__(128, 1)
flash_attn_kernel(const unsigned short* __restrict__ Qb_u,
                  const unsigned short* __restrict__ Kb_u,
                  const unsigned short* __restrict__ Vb_u,
                  const unsigned char* __restrict__ mask,
                  float* __restrict__ Out,
                  int Lq, int Lk, float scale) {
    extern __shared__ char smem_raw[];
    __bf16* vT = (__bf16*)smem_raw;          // 32 * Lk
    __bf16* pS = vT + 32 * Lk;               // 4 waves * 16*32

    const __bf16* Qb = (const __bf16*)Qb_u;
    const __bf16* Kb = (const __bf16*)Kb_u;
    const __bf16* Vb = (const __bf16*)Vb_u;

    int lane = threadIdx.x & 31;
    int warp = threadIdx.x >> 5;
    int qchunks = (Lq + 63) / 64;
    int bx = blockIdx.x;
    int qc = bx % qchunks;
    int h  = (bx / qchunks) & 7;
    int g  = bx / (qchunks * 8);

    // ---- stage V^T into LDS -------------------------------------------------
    const __bf16* vg = Vb + ((size_t)g * Lk) * DD + h * DH;
    for (int r = threadIdx.x; r < Lk; r += blockDim.x) {
        const __bf16* vr = vg + (size_t)r * DD;
#pragma unroll
        for (int c = 0; c < DH; ++c) vT[c * Lk + r] = vr[c];
    }
    __syncthreads();

    int q0 = qc * 64 + warp * 16;
    if (q0 >= Lq) return;
    int rows_valid = (Lq - q0 < 16) ? (Lq - q0) : 16;

    const __bf16* qg = Qb + ((size_t)g * Lq + q0) * DD + h * DH;
    v16bf qf = load_frag_guard(qg, DD, lane, rows_valid);

    const __bf16* kg = Kb + ((size_t)g * Lk) * DD + h * DH;
    const unsigned char* mg = mask ? (mask + (size_t)g * Lk * Lk) : nullptr;
    __bf16* myP = pS + warp * (16 * 32);

    int col   = lane & 15;
    int rbase = (lane & 16) ? 8 : 0;

    float m[8], l[8];
    v8f o0 = {}, o1 = {};
#pragma unroll
    for (int j = 0; j < 8; ++j) { m[j] = -3.0e38f; l[j] = 0.0f; }

    for (int kc = 0; kc < Lk; kc += 32) {
        // prefetch next K chunk (predictable stream; lowers to global_prefetch)
        if (kc + 32 < Lk)
            __builtin_prefetch(kg + (size_t)(kc + 32) * DD + (lane & 15) * DD, 0, 3);

        // scores: S = Q (16x32) @ K^T (32x32) as two 16x16 wmmas
        v16bf b0 = load_frag(kg + (size_t)kc * DD, DD, lane);
        v16bf b1 = load_frag(kg + (size_t)(kc + 16) * DD, DD, lane);
        v8f z = {};
        v8f s0 = wmma_bf16(qf, b0, z);
        v8f s1 = wmma_bf16(qf, b1, z);

#pragma unroll
        for (int j = 0; j < 8; ++j) {
            float x0 = s0[j] * scale;
            float x1 = s1[j] * scale;
            if (mg) {
                int qr = q0 + rbase + j;
                const unsigned char* mr = mg + (size_t)qr * Lk + kc;
                if (!mr[col])      x0 = -1.0e9f;
                if (!mr[col + 16]) x1 = -1.0e9f;
            }
            // row max across the 16 lanes holding this row
            float tm = fmaxf(x0, x1);
            tm = fmaxf(tm, __shfl_xor(tm, 1, 32));
            tm = fmaxf(tm, __shfl_xor(tm, 2, 32));
            tm = fmaxf(tm, __shfl_xor(tm, 4, 32));
            tm = fmaxf(tm, __shfl_xor(tm, 8, 32));
            float mn = fmaxf(m[j], tm);
            float alpha = __expf(m[j] - mn);
            m[j] = mn;
            float p0 = __expf(x0 - mn);
            float p1 = __expf(x1 - mn);
            float rs = p0 + p1;
            rs += __shfl_xor(rs, 1, 32);
            rs += __shfl_xor(rs, 2, 32);
            rs += __shfl_xor(rs, 4, 32);
            rs += __shfl_xor(rs, 8, 32);
            l[j] = l[j] * alpha + rs;
            o0[j] *= alpha;
            o1[j] *= alpha;
            // stage P tile (C-layout scatter -> row-major 16x32 bf16)
            myP[(rbase + j) * 32 + col]      = (__bf16)p0;
            myP[(rbase + j) * 32 + col + 16] = (__bf16)p1;
        }

        // O (16x32) += P (16x32) @ V (32x32): two wmmas via LDS V^T
        v16bf pf  = load_frag(myP, 32, lane);
        v16bf vf0 = load_frag(vT + kc, Lk, lane);
        v16bf vf1 = load_frag(vT + 16 * Lk + kc, Lk, lane);
        o0 = wmma_bf16(pf, vf0, o0);
        o1 = wmma_bf16(pf, vf1, o1);
    }

    float* outg = Out + ((size_t)g * Lq + q0) * DD + h * DH;
#pragma unroll
    for (int j = 0; j < 8; ++j) {
        int r = rbase + j;
        if (r < rows_valid) {
            float inv = 1.0f / l[j];
            outg[(size_t)r * DD + col]      = o0[j] * inv;
            outg[(size_t)r * DD + col + 16] = o1[j] * inv;
        }
    }
}

// ---------------------------------------------------------------------------
// LayerNorm with residual add (row width 256): out = LN(A + B) * g + b
// One wave per row, 4 rows per block.
// ---------------------------------------------------------------------------
__global__ void ln_resid_kernel(const float* __restrict__ A, const float* __restrict__ B,
                                const float* __restrict__ gam, const float* __restrict__ bet,
                                float* __restrict__ outF, unsigned short* __restrict__ outB_u,
                                int rows) {
    __bf16* outB = (__bf16*)outB_u;
    int lane = threadIdx.x & 31;
    int warp = threadIdx.x >> 5;
    int row = blockIdx.x * 4 + warp;
    if (row >= rows) return;
    const float* a = A + (size_t)row * DD;
    const float* b = B ? (B + (size_t)row * DD) : nullptr;
    float v[8];
    float s = 0.0f, sq = 0.0f;
#pragma unroll
    for (int i = 0; i < 8; ++i) {
        int c = lane + i * 32;
        float t = a[c] + (b ? b[c] : 0.0f);
        v[i] = t;
        s += t;
        sq += t * t;
    }
    for (int msk = 1; msk < 32; msk <<= 1) {
        s  += __shfl_xor(s, msk, 32);
        sq += __shfl_xor(sq, msk, 32);
    }
    float mean = s * (1.0f / 256.0f);
    float var  = sq * (1.0f / 256.0f) - mean * mean;
    float rstd = rsqrtf(var + 1e-5f);
#pragma unroll
    for (int i = 0; i < 8; ++i) {
        int c = lane + i * 32;
        float o = (v[i] - mean) * rstd * gam[c] + bet[c];
        if (outF) outF[(size_t)row * DD + c] = o;
        if (outB) outB[(size_t)row * DD + c] = (__bf16)o;
    }
}

// ---------------------------------------------------------------------------
// PMA seed projection:  Qp_pma[g, c] = seed @ Wq + bq   (same for all g)
// ---------------------------------------------------------------------------
__global__ void pma_q_kernel(const float* __restrict__ seed, const float* __restrict__ Wq,
                             const float* __restrict__ bq, float* __restrict__ outF,
                             unsigned short* __restrict__ outB_u) {
    __bf16* outB = (__bf16*)outB_u;
    int idx = blockIdx.x * blockDim.x + threadIdx.x;
    if (idx >= GG * DD) return;
    int c = idx & 255;
    float acc = bq[c];
    for (int k = 0; k < DD; ++k) acc += seed[k] * Wq[(size_t)k * DD + c];
    outF[idx] = acc;
    outB[idx] = (__bf16)acc;
}

// ---------------------------------------------------------------------------
// Classifier head
// ---------------------------------------------------------------------------
__global__ void cls1_kernel(const float* __restrict__ pooled, const float* __restrict__ W1,
                            const float* __restrict__ b1, float* __restrict__ h) {
    int idx = blockIdx.x * blockDim.x + threadIdx.x;
    if (idx >= GG * HIDC) return;
    int g = idx / HIDC, c = idx % HIDC;
    float acc = b1[c];
    for (int k = 0; k < DD; ++k) acc += pooled[(size_t)g * DD + k] * W1[(size_t)k * HIDC + c];
    h[idx] = fmaxf(acc, 0.0f);
}

__global__ void cls2_kernel(const float* __restrict__ h, const float* __restrict__ W2,
                            const float* __restrict__ b2, float* __restrict__ out) {
    int g = threadIdx.x;
    if (g >= GG) return;
    float acc = b2[0];
    for (int k = 0; k < HIDC; ++k) acc += h[(size_t)g * HIDC + k] * W2[k];
    out[g] = acc;
}

// ---------------------------------------------------------------------------
// Host launcher
// ---------------------------------------------------------------------------
extern "C" void kernel_launch(void* const* d_in, const int* in_sizes, int n_in,
                              void* d_out, int out_size, void* d_ws, size_t ws_size,
                              hipStream_t stream) {
    (void)in_sizes; (void)n_in; (void)out_size; (void)ws_size;

    // ---- input map (setup_inputs dict order, recursive insertion order) ----
    const float* x  = (const float*)d_in[0];
    const int*   ei = (const int*)d_in[1];
    const float* ea = (const float*)d_in[2];
    // d_in[3] = batch (unused)
    const float* node_W = (const float*)d_in[4];
    const float* node_b = (const float*)d_in[5];
    const float* edge_W = (const float*)d_in[6];
    const float* edge_b = (const float*)d_in[7];
    // mab params: 12 entries each: Wq,bq,Wk,bk,Wv,bv,Wo,bo,ln0_g,ln0_b,ln1_g,ln1_b
    auto MABP = [&](int i, int j) -> const float* {
        int base = (i < 3) ? (8 + 12 * i) : 44;
        return (const float*)d_in[base + j];
    };
    const float* seed   = (const float*)d_in[56];
    const float* cls_W1 = (const float*)d_in[57];
    const float* cls_b1 = (const float*)d_in[58];
    const float* cls_W2 = (const float*)d_in[59];
    const float* cls_b2 = (const float*)d_in[60];
    float* logits = (float*)d_out;

    // ---- workspace carve ----------------------------------------------------
    char* ws = (char*)d_ws;
    size_t off = 0;
    auto alloc = [&](size_t bytes) -> void* {
        void* p = ws + off;
        off += (bytes + 255) & ~(size_t)255;
        return p;
    };
    unsigned short* WT[4][4];   // [mab][q,k,v,o] : 256x256 bf16 transposed
    for (int i = 0; i < 4; ++i)
        for (int j = 0; j < 4; ++j)
            WT[i][j] = (unsigned short*)alloc((size_t)DD * DD * 2);

    float*          nodef = (float*)alloc((size_t)GG * NN * 128 * 4);
    float*          edgef = (float*)alloc((size_t)NROW * 128 * 4);
    unsigned char*  maskb = (unsigned char*)alloc((size_t)GG * EE * EE);
    unsigned short* erB   = (unsigned short*)alloc((size_t)NROW * DD * 2);
    float*          QPF   = (float*)alloc((size_t)NROW * DD * 4);   // Qp f32, reused as R f32
    unsigned short* QB    = (unsigned short*)alloc((size_t)NROW * DD * 2);
    unsigned short* KB    = (unsigned short*)alloc((size_t)NROW * DD * 2);
    unsigned short* VB    = (unsigned short*)alloc((size_t)NROW * DD * 2);
    float*          OF    = (float*)alloc((size_t)NROW * DD * 4);
    float*          H1F   = (float*)alloc((size_t)NROW * DD * 4);
    unsigned short* H1B   = (unsigned short*)alloc((size_t)NROW * DD * 2);
    float*          pmaQF = (float*)alloc((size_t)GG * DD * 4);
    unsigned short* pmaQB = (unsigned short*)alloc((size_t)GG * DD * 2);
    float*          pmaOF = (float*)alloc((size_t)GG * DD * 4);
    float*          pmaH1 = (float*)alloc((size_t)GG * DD * 4);
    unsigned short* pmaH1B= (unsigned short*)alloc((size_t)GG * DD * 2);
    float*          pmaR  = (float*)alloc((size_t)GG * DD * 4);
    float*          pooled= (float*)alloc((size_t)GG * DD * 4);
    float*          hcls  = (float*)alloc((size_t)GG * HIDC * 4);

    const int TB = 256;

    // ---- weight transposes (fp32 -> bf16, N x K) ----------------------------
    for (int i = 0; i < 4; ++i)
        for (int j = 0; j < 4; ++j)
            transpose_w_kernel<<<(DD * DD + TB - 1) / TB, TB, 0, stream>>>(
                MABP(i, 2 * j), WT[i][j], DD, DD);

    // ---- input projections + edge_repr + mask -------------------------------
    proj_kernel<<<(GG * NN * 128 + TB - 1) / TB, TB, 0, stream>>>(x, node_W, node_b,
                                                                  nodef, GG * NN, 32);
    proj_kernel<<<(NROW * 128 + TB - 1) / TB, TB, 0, stream>>>(ea, edge_W, edge_b,
                                                               edgef, NROW, 16);
    assemble_kernel<<<(NROW * DD + TB - 1) / TB, TB, 0, stream>>>(nodef, edgef, ei, erB);
    mask_kernel<<<GG, 256, 0, stream>>>(ei, maskb);

    const float scale = 1.0f / 16.0f;  // 1/sqrt(D), D=256
    // GEMM task = 16x64 strip; 4 waves/block
    const int gemm_blocks = ((NROW / 16) * (DD / 64) + 3) / 4;   // 1024
    const int ln_blocks   = (NROW + 3) / 4;                      // 4096
    const size_t flash_lds = (size_t)(32 * EE + 4 * 16 * 32) * 2; // 36864 B

    // ---- 3 SAB layers -------------------------------------------------------
    for (int i = 0; i < 3; ++i) {
        gemm_bf16_kernel<<<gemm_blocks, 128, 0, stream>>>(erB, WT[i][0], MABP(i, 1),
                                                          QPF, QB, NROW, DD, DD, 0);
        gemm_bf16_kernel<<<gemm_blocks, 128, 0, stream>>>(erB, WT[i][1], MABP(i, 3),
                                                          nullptr, KB, NROW, DD, DD, 0);
        gemm_bf16_kernel<<<gemm_blocks, 128, 0, stream>>>(erB, WT[i][2], MABP(i, 5),
                                                          nullptr, VB, NROW, DD, DD, 0);
        const unsigned char* mptr = (i < 2) ? maskb : nullptr;
        flash_attn_kernel<<<GG * HEADS * (EE / 64), 128, flash_lds, stream>>>(
            QB, KB, VB, mptr, OF, EE, EE, scale);
        ln_resid_kernel<<<ln_blocks, 128, 0, stream>>>(QPF, OF, MABP(i, 8), MABP(i, 9),
                                                       H1F, H1B, NROW);
        // R = relu(H1 @ Wo + bo)  (reuse QPF)
        gemm_bf16_kernel<<<gemm_blocks, 128, 0, stream>>>(H1B, WT[i][3], MABP(i, 7),
                                                          QPF, nullptr, NROW, DD, DD, 1);
        // er_next = LN(H1 + R)  (only bf16 needed downstream)
        ln_resid_kernel<<<ln_blocks, 128, 0, stream>>>(H1F, QPF, MABP(i, 10), MABP(i, 11),
                                                       nullptr, erB, NROW);
    }

    // ---- PMA pooling --------------------------------------------------------
    pma_q_kernel<<<(GG * DD + TB - 1) / TB, TB, 0, stream>>>(seed, MABP(3, 0), MABP(3, 1),
                                                             pmaQF, pmaQB);
    gemm_bf16_kernel<<<gemm_blocks, 128, 0, stream>>>(erB, WT[3][1], MABP(3, 3),
                                                      nullptr, KB, NROW, DD, DD, 0);
    gemm_bf16_kernel<<<gemm_blocks, 128, 0, stream>>>(erB, WT[3][2], MABP(3, 5),
                                                      nullptr, VB, NROW, DD, DD, 0);
    flash_attn_kernel<<<GG * HEADS, 128, flash_lds, stream>>>(
        pmaQB, KB, VB, nullptr, pmaOF, 1, EE, scale);
    ln_resid_kernel<<<(GG + 3) / 4, 128, 0, stream>>>(pmaQF, pmaOF, MABP(3, 8), MABP(3, 9),
                                                      pmaH1, pmaH1B, GG);
    {
        const int small_blocks = ((GG / 16) * (DD / 64) + 3) / 4;  // 2
        gemm_bf16_kernel<<<small_blocks, 128, 0, stream>>>(pmaH1B, WT[3][3], MABP(3, 7),
                                                           pmaR, nullptr, GG, DD, DD, 1);
    }
    ln_resid_kernel<<<(GG + 3) / 4, 128, 0, stream>>>(pmaH1, pmaR, MABP(3, 10), MABP(3, 11),
                                                      pooled, nullptr, GG);

    // ---- classifier ---------------------------------------------------------
    cls1_kernel<<<(GG * HIDC + TB - 1) / TB, TB, 0, stream>>>(pooled, cls_W1, cls_b1, hcls);
    cls2_kernel<<<1, 32, 0, stream>>>(hcls, cls_W2, cls_b2, logits);
}
